// BIOTEncoder_84473416777891
// MI455X (gfx1250) — matrix operand order repaired and checked by reading, not compiled, not run on hardware
//
#include <hip/hip_runtime.h>
#include <math.h>

typedef __bf16 bf16;
typedef __attribute__((ext_vector_type(16))) __bf16 v16bf;
typedef __attribute__((ext_vector_type(8)))  float  v8f;
typedef __attribute__((ext_vector_type(4)))  unsigned int v4u;
typedef __attribute__((ext_vector_type(4)))  int v4i_t;
typedef __attribute__((ext_vector_type(8)))  int v8i_t;

#define WMMA_BF16(a, b, c) \
  __builtin_amdgcn_wmma_f32_16x16x32_bf16(false, (a), false, (b), (short)0, (c), false, false)

#if __has_builtin(__builtin_amdgcn_tensor_load_to_lds) && __has_builtin(__builtin_amdgcn_s_wait_tensorcnt)
#define HAVE_TDM 1
#if __has_include(<hip/amd_detail/amd_gfx1250_TDM.h>)
#define TDM_CALL(g0, g1, g2, g3) \
  __builtin_amdgcn_tensor_load_to_lds((g0), (g1), (g2), (g3), (v8i_t){0, 0, 0, 0, 0, 0, 0, 0}, 0)
#else
#define TDM_CALL(g0, g1, g2, g3) __builtin_amdgcn_tensor_load_to_lds((g0), (g1), (g2), (g3), 0)
#endif
#else
#define HAVE_TDM 0
#endif

// ---------------- problem constants ----------------
constexpr int B_    = 128;
constexpr int C_    = 23;
constexpr int T_    = 2000;
constexpr int NFFT  = 200;
constexpr int HOP   = 100;
constexpr int FT_   = 19;
constexpr int NFREQ = 101;
constexpr int EMB   = 256;
constexpr int HEADS = 8;
constexpr int DH    = 32;
constexpr int FF    = 1024;
constexpr int DEPTH = 4;
constexpr int S_    = C_ * FT_;        // 437
constexpr int R_    = B_ * S_;         // 55936 = 437*128

// ---------------- WMMA fragment loaders (CDNA5 ISA 7.12.2 layouts, b128) ----------------
__device__ __forceinline__ v16bf frag_a(const bf16* base, int lane) {
  const bf16* p = base + ((lane >> 4) << 3);
  union { v16bf v; uint4 q[2]; } u;
  u.q[0] = *(const uint4*)p;
  u.q[1] = *(const uint4*)(p + 16);
  return u.v;
}
__device__ __forceinline__ v16bf frag_b(const bf16* rowbase, int lane) {
  const bf16* p = rowbase + ((lane >> 4) << 3);
  union { v16bf v; uint4 q[2]; } u;
  u.q[0] = *(const uint4*)p;
  u.q[1] = *(const uint4*)(p + 16);
  return u.v;
}

#if HAVE_TDM
__device__ __forceinline__ unsigned lds_off(const void* p) { return (unsigned)(size_t)p; }
// 2D TDM load: tile (w elems x h rows) of 2-byte data, row stride = stride elems.
__device__ __forceinline__ void tdm_load_2d(unsigned lds_byte, const void* gptr,
                                            unsigned w, unsigned h, unsigned stride) {
  unsigned long long ga = (unsigned long long)(size_t)gptr;
  v4u g0 = {0u, 0u, 0u, 0u};
  g0[0] = 1u;                                             // count = 1
  g0[1] = lds_byte;                                       // lds_addr
  g0[2] = (unsigned)ga;                                   // global_addr lo
  g0[3] = (unsigned)((ga >> 32) & 0x01FFFFFFull) | 0x80000000u;  // addr hi | type=2
  v8i_t g1 = {0, 0, 0, 0, 0, 0, 0, 0};
  g1[0] = (int)(1u << 16);                                // data_size = 2 bytes
  g1[1] = (int)((w & 0xFFFFu) << 16);                     // tensor_dim0 lo16
  g1[2] = (int)((w >> 16) | ((h & 0xFFFFu) << 16));       // dim0 hi | tensor_dim1 lo16
  g1[3] = (int)((h >> 16) | ((w & 0xFFFFu) << 16));       // dim1 hi | tile_dim0
  g1[4] = (int)(h & 0xFFFFu);                             // tile_dim1 (tile_dim2 = 0)
  g1[5] = (int)stride;                                    // tensor_dim0_stride lo32
  v4i_t z4 = {0, 0, 0, 0};
  TDM_CALL(g0, g1, z4, z4);
}
#endif

// ---------------- weight prep: fp32 [mat][K][N] -> bf16 transposed [mat][N][K] ----------------
__global__ void k_transp(const float* __restrict__ in, bf16* __restrict__ out,
                         int K, int N, int total) {
  for (int i = blockIdx.x * blockDim.x + threadIdx.x; i < total; i += gridDim.x * blockDim.x) {
    int mat = i / (K * N), r = i % (K * N);
    int n = r / K, k = r % K;
    out[i] = (bf16)in[(size_t)mat * K * N + (size_t)k * N + n];
  }
}

// W_proj (101x256) -> transposed zero-padded bf16 [256][128]
__global__ void k_prep_wpT(const float* __restrict__ wp, bf16* __restrict__ out) {
  int i = blockIdx.x * blockDim.x + threadIdx.x;
  if (i >= 256 * 128) return;
  int n = i >> 7, k = i & 127;
  out[i] = (k < NFREQ) ? (bf16)wp[k * 256 + n] : (bf16)0.0f;
}

// DFT basis, n-major: bas[n][k] for n<112, k<224 (zero for k>=200)
__global__ void k_prep_basis(bf16* __restrict__ basC, bf16* __restrict__ basS) {
  int i = blockIdx.x * blockDim.x + threadIdx.x;
  if (i >= 112 * 224) return;
  int n = i / 224, k = i % 224;
  if (k < NFFT) {
    float ang = 6.283185307f * (float)(n * k) / (float)NFFT;
    basC[i] = (bf16)cosf(ang);
    basS[i] = (bf16)sinf(ang);
  } else {
    basC[i] = (bf16)0.0f;
    basS[i] = (bf16)0.0f;
  }
}

// ---------------- STFT + embedding (WMMA DFT + WMMA projection) ----------------
__global__ void k_stft_embed(const float* __restrict__ x, const unsigned char* __restrict__ mask,
                             const int* __restrict__ offp, const bf16* __restrict__ wpT,
                             const bf16* __restrict__ basC, const bf16* __restrict__ basS,
                             const float* __restrict__ bproj, const float* __restrict__ chtok,
                             float* __restrict__ emb_c, float* __restrict__ h_out) {
  extern __shared__ bf16 sm[];
  bf16* frames = sm;                    // [2][32][224] zero-padded DFT A tiles
  bf16* spec   = frames + 2 * 32 * 224; // [2][32][128]
  bf16* wpt    = spec + 2 * 32 * 128;   // [256][128] transposed W_proj

  const int t = threadIdx.x, lane = t & 31, wave = t >> 5;
  const int b = blockIdx.x / C_, c = blockIdx.x % C_;
  const size_t xb = ((size_t)b * C_ + c) * T_;

#if HAVE_TDM
  if (wave == 0) {
    tdm_load_2d(lds_off(wpt), wpT, 128, 256, 128);
    __builtin_amdgcn_s_wait_tensorcnt(0);
  }
#else
  for (int lin8 = t; lin8 < 4096; lin8 += 256)
    *(uint4*)&wpt[lin8 * 8] = *(const uint4*)(wpT + (size_t)lin8 * 8);
#endif

  // branchless frames fill: pairs of elements, clamp-index global loads
  for (int p = t; p < 3584; p += 256) {          // 32*224/2 pairs per signal
    int m = p / 112, k2 = (p % 112) * 2;
    bool ok = (m < FT_) && (k2 < NFFT);
    int idx = ok ? (m * HOP + k2) : 0;
    float2 xv = *(const float2*)(x + xb + idx);
    unsigned char mk0 = mask[xb + idx], mk1 = mask[xb + idx + 1];
    float a0 = ok ? xv.x : 0.0f, a1 = ok ? xv.y : 0.0f;
    union { unsigned u; bf16 h[2]; } uc, um;
    uc.h[0] = (bf16)a0;                uc.h[1] = (bf16)a1;
    um.h[0] = (bf16)(mk0 ? 0.0f : a0); um.h[1] = (bf16)(mk1 ? 0.0f : a1);
    ((unsigned*)frames)[p]        = uc.u;
    ((unsigned*)frames)[3584 + p] = um.u;
  }
  __syncthreads();

  // ---- DFT via WMMA; basis read as b128 from global (L2 resident, shared by all WGs) ----
  for (int task = wave; task < 28; task += 8) {
    int sg = task & 1, mt = (task >> 1) & 1, nt = task >> 2;   // nt 0..6
    int m = mt * 16 + (lane & 15);
    int n = nt * 16 + (lane & 15);
    v8f ar = {}, ai = {};
    for (int kc = 0; kc < 7; ++kc) {
      v16bf af = frag_a(frames + sg * 7168 + m * 224 + kc * 32, lane);
      v16bf bc = frag_b(basC + n * 224 + kc * 32, lane);
      v16bf bs = frag_b(basS + n * 224 + kc * 32, lane);
      ar = WMMA_BF16(af, bc, ar);
      ai = WMMA_BF16(af, bs, ai);
    }
#pragma unroll
    for (int r = 0; r < 8; ++r) {
      int mm = mt * 16 + ((lane >> 4) << 3) + r;
      float re = ar[r], im = ai[r];
      spec[sg * 4096 + mm * 128 + n] = (bf16)sqrtf(re * re + im * im);
    }
  }
  __syncthreads();
  for (int i = t; i < 2 * 32 * 128; i += 256)   // zero K-pad cols >= NFREQ
    if ((i & 127) >= NFREQ) spec[i] = (bf16)0.0f;
  __syncthreads();

  // ---- projection + fused epilogue (bias + channel token + positional encoding) ----
  const int off = *offp;
  int ci = c + off; if (ci < 0) ci = 0; if (ci >= C_) ci = C_ - 1;
  const float LOG1E4_D = logf(10000.0f) / (float)EMB;
  for (int task = wave; task < 64; task += 8) {
    int sg = task & 1, mt = (task >> 1) & 1, nt = task >> 2;   // nt 0..15
    int m = mt * 16 + (lane & 15);
    int n = nt * 16 + (lane & 15);
    v8f acc = {};
    for (int kc = 0; kc < 4; ++kc) {
      v16bf af = frag_a(spec + sg * 4096 + m * 128 + kc * 32, lane);
      v16bf bf = frag_b(wpt + n * 128 + kc * 32, lane);
      acc = WMMA_BF16(af, bf, acc);
    }
    float div = expf(-(float)(n & ~1) * LOG1E4_D);
    float add = bproj[n] + chtok[ci * EMB + n];
#pragma unroll
    for (int r = 0; r < 8; ++r) {
      int tf = mt * 16 + ((lane >> 4) << 3) + r;
      if (tf >= FT_) continue;
      float pe  = (n & 1) ? cosf((float)tf * div) : sinf((float)tf * div);
      float val = acc[r] + add + pe;
      size_t row = (size_t)b * S_ + c * FT_ + tf;
      if (sg == 0) emb_c[row * EMB + n] = val;
      else         h_out[row * EMB + n] = val;
    }
  }
}

// ---------------- LayerNorm -> bf16 (wave per row) ----------------
__global__ void k_ln(const float* __restrict__ h, const float* __restrict__ g,
                     const float* __restrict__ bt, bf16* __restrict__ out, int rows) {
  int lane = threadIdx.x & 31, wave = threadIdx.x >> 5;
  int row = blockIdx.x * 8 + wave;
  if (row >= rows) return;
  const float* p = h + (size_t)row * EMB + lane * 8;
  float v[8];
  *(float4*)&v[0] = *(const float4*)p;
  *(float4*)&v[4] = *(const float4*)(p + 4);
  float s = 0.f, q = 0.f;
#pragma unroll
  for (int j = 0; j < 8; ++j) { s += v[j]; q += v[j] * v[j]; }
#pragma unroll
  for (int o = 16; o; o >>= 1) { s += __shfl_xor(s, o, 32); q += __shfl_xor(q, o, 32); }
  float mean = s / EMB;
  float rstd = rsqrtf(q / EMB - mean * mean + 1e-5f);
  float gg[8], bb[8];
  *(float4*)&gg[0] = *(const float4*)(g + lane * 8);
  *(float4*)&gg[4] = *(const float4*)(g + lane * 8 + 4);
  *(float4*)&bb[0] = *(const float4*)(bt + lane * 8);
  *(float4*)&bb[4] = *(const float4*)(bt + lane * 8 + 4);
  union { uint4 q4; bf16 h8[8]; } u;
#pragma unroll
  for (int j = 0; j < 8; ++j) u.h8[j] = (bf16)((v[j] - mean) * rstd * gg[j] + bb[j]);
  *(uint4*)(out + (size_t)row * EMB + lane * 8) = u.q4;
}

// ---------------- softmax over feature dim (q), fused *scale -> bf16 ----------------
__global__ void k_softmax_feat(const float* __restrict__ qf, bf16* __restrict__ qb, int rows) {
  int idx = blockIdx.x * blockDim.x + threadIdx.x;  // row*8 + head
  if (idx >= rows * HEADS) return;
  size_t base = (size_t)(idx >> 3) * EMB + (size_t)(idx & 7) * DH;
  float v[DH];
  const float4* p4 = (const float4*)(qf + base);
#pragma unroll
  for (int j = 0; j < 8; ++j) *(float4*)&v[4 * j] = p4[j];
  float m = v[0];
#pragma unroll
  for (int d = 1; d < DH; ++d) m = fmaxf(m, v[d]);
  float ssum = 0.f;
#pragma unroll
  for (int d = 0; d < DH; ++d) { v[d] = expf(v[d] - m); ssum += v[d]; }
  float inv = 0.1767766953f / ssum;   // scale = DH^-0.5
#pragma unroll
  for (int gr = 0; gr < 4; ++gr) {
    union { uint4 q4; bf16 h8[8]; } u;
#pragma unroll
    for (int j = 0; j < 8; ++j) u.h8[j] = (bf16)(v[gr * 8 + j] * inv);
    *(uint4*)(qb + base + gr * 8) = u.q4;
  }
}

// ---------------- softmax over sequence dim (k) ----------------
__global__ void k_softmax_seq(const float* __restrict__ kf, bf16* __restrict__ kb) {
  int col = blockIdx.x * blockDim.x + threadIdx.x;  // b*EMB + f
  if (col >= B_ * EMB) return;
  size_t base = (size_t)(col >> 8) * S_ * EMB + (col & 255);
  float m = -3.4e38f;
  for (int s = 0; s < S_; ++s) m = fmaxf(m, kf[base + (size_t)s * EMB]);
  float ssum = 0.f;
  for (int s = 0; s < S_; ++s) ssum += expf(kf[base + (size_t)s * EMB] - m);
  float inv = 1.0f / ssum;
  for (int s = 0; s < S_; ++s)
    kb[base + (size_t)s * EMB] = (bf16)(expf(kf[base + (size_t)s * EMB] - m) * inv);
}

// ---------------- ctx = K^T @ V per (b,h): 32x32, K = 448 (padded S) ----------------
// ctx stored TRANSPOSED in global: ctx[(bh*32 + e)*32 + d]
__global__ void k_ctx(const bf16* __restrict__ kb, const bf16* __restrict__ vb,
                      bf16* __restrict__ ctxg) {
  extern __shared__ bf16 sm[];
  bf16* kT = sm;            // [32 d][448 s]
  bf16* vT = sm + 32 * 448; // [32 e][448 s]
  int t = threadIdx.x, lane = t & 31, wave = t >> 5;
  int b = blockIdx.x >> 3, hh = blockIdx.x & 7;
  for (int lin8 = t; lin8 < 1792; lin8 += 256) {   // 448*32/8
    int s = lin8 >> 2, d8 = (lin8 & 3) * 8;
    union { uint4 q4; bf16 h8[8]; } uk, uv;
    uint4 z = {0, 0, 0, 0};
    uk.q4 = z; uv.q4 = z;
    if (s < S_) {
      size_t g = ((size_t)b * S_ + s) * EMB + hh * DH + d8;
      uk.q4 = *(const uint4*)(kb + g);
      uv.q4 = *(const uint4*)(vb + g);
    }
#pragma unroll
    for (int j = 0; j < 8; ++j) {
      kT[(d8 + j) * 448 + s] = uk.h8[j];
      vT[(d8 + j) * 448 + s] = uv.h8[j];
    }
  }
  __syncthreads();
  if (wave < 4) {
    int mt = wave & 1, nt = wave >> 1;
    int m = mt * 16 + (lane & 15);
    int n = nt * 16 + (lane & 15);
    v8f acc = {};
    for (int kc = 0; kc < 14; ++kc) {
      v16bf af = frag_a(kT + m * 448 + kc * 32, lane);
      v16bf bf = frag_b(vT + n * 448 + kc * 32, lane);
      acc = WMMA_BF16(af, bf, acc);
    }
    union { uint4 q4; bf16 h8[8]; } u;
#pragma unroll
    for (int r = 0; r < 8; ++r) u.h8[r] = (bf16)acc[r];
    *(uint4*)(ctxg + ((size_t)blockIdx.x * 32 + n) * 32 + mt * 16 + ((lane >> 4) << 3)) = u.q4;
  }
}

// ---------------- attn = q @ ctx per (b,h): S x 32, K = 32 ----------------
__global__ void k_attn(const bf16* __restrict__ qb, const bf16* __restrict__ ctxg,
                       bf16* __restrict__ attnb) {
  extern __shared__ bf16 sm[];
  bf16* qS   = sm;            // [448 s][32 d]
  bf16* ctxT = sm + 448 * 32; // [32 e][32 d]
  int t = threadIdx.x, lane = t & 31, wave = t >> 5;
  int b = blockIdx.x >> 3, hh = blockIdx.x & 7;
  for (int lin8 = t; lin8 < 1792; lin8 += 256) {
    int s = lin8 >> 2, d8 = (lin8 & 3) * 8;
    uint4 q = {0, 0, 0, 0};
    if (s < S_) q = *(const uint4*)(qb + ((size_t)b * S_ + s) * EMB + hh * DH + d8);
    *(uint4*)&qS[s * 32 + d8] = q;
  }
  for (int lin8 = t; lin8 < 128; lin8 += 256)
    *(uint4*)&ctxT[lin8 * 8] = *(const uint4*)(ctxg + (size_t)blockIdx.x * 1024 + (size_t)lin8 * 8);
  __syncthreads();
  for (int task = wave; task < 56; task += 8) {
    int mt = task >> 1, nt = task & 1;
    int m = mt * 16 + (lane & 15);
    int n = nt * 16 + (lane & 15);
    v16bf af = frag_a(qS + m * 32, lane);
    v16bf bf = frag_b(ctxT + n * 32, lane);
    v8f acc = {};
    acc = WMMA_BF16(af, bf, acc);
#pragma unroll
    for (int r = 0; r < 8; ++r) {
      int s = mt * 16 + ((lane >> 4) << 3) + r;
      if (s >= S_) continue;
      attnb[((size_t)b * S_ + s) * EMB + hh * DH + n] = (bf16)acc[r];
    }
  }
}

// ---------------- generic bf16 WMMA GEMM: out(M,N) = A(M,K) @ W(K,N), W pre-transposed [N][K] ----
// 128x64 block tile, 8 waves, each wave 32x32 output = 4 WMMA per K-chunk with frag reuse.
// mode: 0 = f32 store, 1 = f32 residual accumulate (+bias), 2 = bf16 store, 3 = gelu->bf16 (+bias)
// M % 128 == 0, N % 64 == 0, K % 32 == 0 (guaranteed by caller)
__global__ void k_gemm(const bf16* __restrict__ A, const bf16* __restrict__ WT,
                       const float* __restrict__ bias, void* __restrict__ out,
                       int M, int N, int K, int mode) {
  extern __shared__ bf16 sm[];
  bf16* At = sm;        // [128 m][32 k]
  bf16* Wt = sm + 4096; // [64 n][32 k]
  int t = threadIdx.x, lane = t & 31, wave = t >> 5;
  int wm = wave & 3, wn = wave >> 2;      // 4 m-groups x 2 n-groups of 32
  int m0 = blockIdx.x * 128, n0 = blockIdx.y * 64;
  int nk = K >> 5;

  v8f acc[2][2] = {};
  for (int kc = 0; kc < nk; ++kc) {
#if HAVE_TDM
    if (wave == 0) {
      tdm_load_2d(lds_off(At), A + (size_t)m0 * K + kc * 32, 32, 128, (unsigned)K);
      tdm_load_2d(lds_off(Wt), WT + (size_t)n0 * K + kc * 32, 32, 64, (unsigned)K);
      __builtin_amdgcn_s_wait_tensorcnt(0);
    }
#else
#pragma unroll
    for (int i = 0; i < 2; ++i) {         // A tile: 128x32 = 512 uint4
      int lin = t + 256 * i, row = lin >> 2, c8 = (lin & 3) * 8;
      *(uint4*)&At[row * 32 + c8] = *(const uint4*)(A + (size_t)(m0 + row) * K + kc * 32 + c8);
    }
    {                                      // W tile: 64x32 = 256 uint4
      int row = t >> 2, c8 = (t & 3) * 8;
      *(uint4*)&Wt[row * 32 + c8] = *(const uint4*)(WT + (size_t)(n0 + row) * K + kc * 32 + c8);
    }
    if (kc + 1 < nk)
      __builtin_prefetch(A + (size_t)(m0 + (t >> 2)) * K + (kc + 1) * 32 + (t & 3) * 8, 0, 1);
#endif
    __syncthreads();
    v16bf a0 = frag_a(At + (wm * 32 + (lane & 15)) * 32, lane);
    v16bf a1 = frag_a(At + (wm * 32 + 16 + (lane & 15)) * 32, lane);
    v16bf b0 = frag_b(Wt + (wn * 32 + (lane & 15)) * 32, lane);
    v16bf b1 = frag_b(Wt + (wn * 32 + 16 + (lane & 15)) * 32, lane);
    acc[0][0] = WMMA_BF16(a0, b0, acc[0][0]);
    acc[0][1] = WMMA_BF16(a0, b1, acc[0][1]);
    acc[1][0] = WMMA_BF16(a1, b0, acc[1][0]);
    acc[1][1] = WMMA_BF16(a1, b1, acc[1][1]);
    __syncthreads();
  }

#pragma unroll
  for (int ms = 0; ms < 2; ++ms) {
#pragma unroll
    for (int ns = 0; ns < 2; ++ns) {
      int n = n0 + wn * 32 + ns * 16 + (lane & 15);
      float bv = bias ? bias[n] : 0.0f;
#pragma unroll
      for (int r = 0; r < 8; ++r) {
        int m = m0 + wm * 32 + ms * 16 + ((lane >> 4) << 3) + r;
        size_t idx = (size_t)m * N + n;
        float val = acc[ms][ns][r] + bv;
        if (mode == 0)      ((float*)out)[idx] = val;
        else if (mode == 1) ((float*)out)[idx] += val;
        else if (mode == 2) ((bf16*)out)[idx] = (bf16)val;
        else {
          float x3 = val * val * val;
          float tg = tanhf(0.7978845608f * (val + 0.044715f * x3));
          ((bf16*)out)[idx] = (bf16)(0.5f * val * (1.0f + tg));
        }
      }
    }
  }
}

// ---------------- host orchestration ----------------
extern "C" void kernel_launch(void* const* d_in, const int* in_sizes, int n_in,
                              void* d_out, int out_size, void* d_ws, size_t ws_size,
                              hipStream_t stream) {
  (void)in_sizes; (void)n_in; (void)out_size; (void)ws_size;
  const float* x      = (const float*)d_in[0];
  const unsigned char* mask = (const unsigned char*)d_in[1];
  const int*   noff   = (const int*)d_in[2];
  const float* Wproj  = (const float*)d_in[3];
  const float* bproj  = (const float*)d_in[4];
  const float* chtok  = (const float*)d_in[5];
  const float* ln1g   = (const float*)d_in[6];
  const float* ln1b   = (const float*)d_in[7];
  const float* Wq     = (const float*)d_in[8];
  const float* Wk     = (const float*)d_in[9];
  const float* Wv     = (const float*)d_in[10];
  const float* Wo     = (const float*)d_in[11];
  const float* ln2g   = (const float*)d_in[12];
  const float* ln2b   = (const float*)d_in[13];
  const float* W1     = (const float*)d_in[14];
  const float* b1     = (const float*)d_in[15];
  const float* W2     = (const float*)d_in[16];
  const float* b2     = (const float*)d_in[17];

  char* ws = (char*)d_ws;
  size_t off = 0;
  auto take = [&](size_t bytes) -> char* {
    char* p = ws + off;
    off = (off + bytes + 255) & ~(size_t)255;
    return p;
  };
  bf16*  wpT   = (bf16*)take((size_t)256 * 128 * 2);
  bf16*  basC  = (bf16*)take((size_t)112 * 224 * 2);
  bf16*  basS  = (bf16*)take((size_t)112 * 224 * 2);
  bf16*  wqT   = (bf16*)take((size_t)DEPTH * EMB * EMB * 2);
  bf16*  wkT   = (bf16*)take((size_t)DEPTH * EMB * EMB * 2);
  bf16*  wvT   = (bf16*)take((size_t)DEPTH * EMB * EMB * 2);
  bf16*  woT   = (bf16*)take((size_t)DEPTH * EMB * EMB * 2);
  bf16*  w1T   = (bf16*)take((size_t)DEPTH * EMB * FF * 2);   // [l][n=FF][k=EMB]
  bf16*  w2T   = (bf16*)take((size_t)DEPTH * FF * EMB * 2);   // [l][n=EMB][k=FF]
  float* h     = (float*)take((size_t)R_ * EMB * 4);
  bf16*  hn    = (bf16*)take((size_t)R_ * EMB * 2);
  float* qf    = (float*)take((size_t)R_ * EMB * 4);
  float* kf    = (float*)take((size_t)R_ * EMB * 4);
  bf16*  qb    = (bf16*)take((size_t)R_ * EMB * 2);
  bf16*  kb    = (bf16*)take((size_t)R_ * EMB * 2);
  bf16*  vb    = (bf16*)take((size_t)R_ * EMB * 2);
  bf16*  attnb = (bf16*)take((size_t)R_ * EMB * 2);
  bf16*  ctx   = (bf16*)take((size_t)B_ * HEADS * DH * DH * 2);
  bf16*  gB    = (bf16*)qf;   // FFN hidden aliases qf+kf region (same bytes, disjoint lifetime)

  // weight prep (transpose to [N][K] bf16)
  k_transp<<<512, 256, 0, stream>>>(Wq, wqT, EMB, EMB, DEPTH * EMB * EMB);
  k_transp<<<512, 256, 0, stream>>>(Wk, wkT, EMB, EMB, DEPTH * EMB * EMB);
  k_transp<<<512, 256, 0, stream>>>(Wv, wvT, EMB, EMB, DEPTH * EMB * EMB);
  k_transp<<<512, 256, 0, stream>>>(Wo, woT, EMB, EMB, DEPTH * EMB * EMB);
  k_transp<<<1024, 256, 0, stream>>>(W1, w1T, EMB, FF, DEPTH * EMB * FF);
  k_transp<<<1024, 256, 0, stream>>>(W2, w2T, FF, EMB, DEPTH * FF * EMB);
  k_prep_wpT<<<128, 256, 0, stream>>>(Wproj, wpT);
  k_prep_basis<<<98, 256, 0, stream>>>(basC, basS);

  // STFT magnitude + projection + embedding
  k_stft_embed<<<B_ * C_, 256, (2 * 32 * 224 + 2 * 32 * 128 + 256 * 128) * 2, stream>>>(
      x, mask, noff, wpT, basC, basS, bproj, chtok, (float*)d_out, h);

  dim3 gE(R_ / 128, EMB / 64);
  dim3 gF(R_ / 128, FF / 64);
  for (int l = 0; l < DEPTH; ++l) {
    k_ln<<<R_ / 8, 256, 0, stream>>>(h, ln1g + l * EMB, ln1b + l * EMB, hn, R_);
    k_gemm<<<gE, 256, 12288, stream>>>(hn, wqT + (size_t)l * EMB * EMB, nullptr, qf, R_, EMB, EMB, 0);
    k_gemm<<<gE, 256, 12288, stream>>>(hn, wkT + (size_t)l * EMB * EMB, nullptr, kf, R_, EMB, EMB, 0);
    k_gemm<<<gE, 256, 12288, stream>>>(hn, wvT + (size_t)l * EMB * EMB, nullptr, vb, R_, EMB, EMB, 2);
    k_softmax_feat<<<(R_ * HEADS + 255) / 256, 256, 0, stream>>>(qf, qb, R_);
    k_softmax_seq<<<(B_ * EMB) / 256, 256, 0, stream>>>(kf, kb);
    k_ctx<<<B_ * HEADS, 256, (2 * 32 * 448) * 2, stream>>>(kb, vb, ctx);
    k_attn<<<B_ * HEADS, 256, (448 * 32 + 32 * 32) * 2, stream>>>(qb, ctx, attnb);
    k_gemm<<<gE, 256, 12288, stream>>>(attnb, woT + (size_t)l * EMB * EMB, nullptr, h, R_, EMB, EMB, 1);
    k_ln<<<R_ / 8, 256, 0, stream>>>(h, ln2g + l * EMB, ln2b + l * EMB, hn, R_);
    k_gemm<<<gF, 256, 12288, stream>>>(hn, w1T + (size_t)l * EMB * FF, b1 + l * FF, gB, R_, FF, EMB, 3);
    k_gemm<<<gE, 256, 12288, stream>>>(gB, w2T + (size_t)l * FF * EMB, b2 + l * EMB, h, R_, EMB, FF, 1);
  }

  hipMemcpyAsync((float*)d_out + (size_t)R_ * EMB, h, (size_t)R_ * EMB * sizeof(float),
                 hipMemcpyDeviceToDevice, stream);
}
